// LineLayer_2121713845189
// MI455X (gfx1250) — matrix operand order repaired and checked by reading, not compiled, not run on hardware
//
#include <hip/hip_runtime.h>

// Problem constants (match reference)
#define DB 4
#define DD 128      // D
#define DN 16384    // N junctions
#define DE 65536    // E = 2*lines
#define K1 384      // 3D
#define M1 256      // 2D
#define M2 128      // D
#define COLS 64     // E-columns per block
#define MSG_LD 392  // 384 + 8 pad (halves)
#define H_LD 264    // 256 + 8 pad (halves)
#define TPB 256     // 8 waves (wave32)

typedef __attribute__((ext_vector_type(16))) _Float16 v16h;
typedef __attribute__((ext_vector_type(8)))  float    v8f;

union Frag { v16h v; unsigned int u[8]; };

// ---------------- prep: f32 weights -> f16, fold b1 into BN shift ----------
__global__ void prep_kernel(const float* __restrict__ w1, const float* __restrict__ b1,
                            const float* __restrict__ g,  const float* __restrict__ beta,
                            const float* __restrict__ mu, const float* __restrict__ var,
                            const float* __restrict__ w2,
                            _Float16* __restrict__ w1h, _Float16* __restrict__ w2h,
                            float* __restrict__ scale, float* __restrict__ shift) {
  int i = blockIdx.x * blockDim.x + threadIdx.x;
  int stride = gridDim.x * blockDim.x;
  for (int t = i; t < M1 * K1; t += stride) w1h[t] = (_Float16)w1[t];
  for (int t = i; t < M2 * M1; t += stride) w2h[t] = (_Float16)w2[t];
  for (int t = i; t < M1; t += stride) {
    float s = g[t] * rsqrtf(var[t] + 1e-5f);
    scale[t] = s;
    shift[t] = beta[t] + (b1[t] - mu[t]) * s;
  }
}

__global__ void zero_counts(unsigned int* __restrict__ counts) {
  int i = blockIdx.x * blockDim.x + threadIdx.x;
  if (i < 2 * DB * DN) counts[i] = 0u;
}

__global__ void count_kernel(const int* __restrict__ idx, unsigned int* __restrict__ counts) {
  int i = blockIdx.x * blockDim.x + threadIdx.x;
  if (i < DB * DE) {
    int b = i / DE;
    atomicAdd(&counts[b * DN + idx[i]], 1u);
  }
}

__global__ void init_out(const float* __restrict__ l0, const float* __restrict__ l1,
                         float* __restrict__ out) {
  long long i = (long long)blockIdx.x * blockDim.x + threadIdx.x;
  const long long tot = (long long)DB * DD * DN;
  if (i < tot) { out[i] = l0[i]; out[tot + i] = l1[i]; }
}

// ---------------- fused gather -> MLP(WMMA) -> scaled scatter-add ----------
// Wave mapping: wave w owns row-tiles {2w, 2w+1} (GEMM1) / {w} (GEMM2) and
// loops over all 4 column-tiles, so each W1/W2 element is fetched from L2
// exactly once per block; B-fragments are refetched cheaply from LDS.
__global__ __launch_bounds__(TPB) void line_mp_kernel(
    const float* __restrict__ ldesc, const float* __restrict__ line_enc,
    const int* __restrict__ idx, const unsigned int* __restrict__ counts,
    const _Float16* __restrict__ w1h, const _Float16* __restrict__ w2h,
    const float* __restrict__ scale, const float* __restrict__ shift,
    const float* __restrict__ b2, float* __restrict__ out) {
  __shared__ __align__(16) _Float16 msg[COLS * MSG_LD];  // [col][3D] f16 message
  __shared__ __align__(16) _Float16 hbuf[COLS * H_LD];   // [col][2D] f16 hidden
  __shared__ int   jidx[COLS];
  __shared__ float cinv[COLS];

  const int tid = threadIdx.x;
  const int b   = blockIdx.y;
  const int e0  = blockIdx.x * COLS;

  // Phase 0: junction ids + 1/count for this tile
  for (int c = tid; c < COLS; c += TPB) {
    int j = idx[b * DE + e0 + c];
    jidx[c] = j;
    unsigned int cnt = counts[b * DN + j];
    cinv[c] = 1.0f / fmaxf((float)cnt, 1.0f);
  }
  __syncthreads();

  // Phase 1: gather + f32->f16 message. Pair partner of column c is c^1
  // (tile is 64-aligned, so e^1 stays in-tile).
  for (int i = tid; i < COLS * DD; i += TPB) {
    int c = i & (COLS - 1);
    int d = i >> 6;                       // log2(COLS)
    float v0 = ldesc[((size_t)b * DD + d) * DN + jidx[c]];
    float v1 = ldesc[((size_t)b * DD + d) * DN + jidx[c ^ 1]];
    float v2 = line_enc[((size_t)b * DD + d) * DE + e0 + c];
    msg[c * MSG_LD + d]          = (_Float16)v0;
    msg[c * MSG_LD + DD + d]     = (_Float16)v1;
    msg[c * MSG_LD + 2 * DD + d] = (_Float16)v2;
  }
  __syncthreads();

  const int lane = tid & 31;
  const int wv   = tid >> 5;      // 0..7
  const int l16  = lane & 15;
  const int hh   = lane >> 4;     // lane half (0/1)

  // ISA 7.12.2 A-matrix (16x32 f16) per-VGPR K offsets:
  // lanes 0-15: V0-3 -> K 0..7, V4-7 -> K 16..23 ; lanes 16-31: +8 / +8
  int kA[8];
  #pragma unroll
  for (int v = 0; v < 8; ++v)
    kA[v] = (v < 4) ? (8 * hh + 2 * v) : (16 + 8 * hh + 2 * (v - 4));
  // B-matrix (32x16 f16): same per-VGPR K offsets as A (kA)
  const int kB0 = 16 * hh;

  const v8f zero8 = {0.f, 0.f, 0.f, 0.f, 0.f, 0.f, 0.f, 0.f};

  // ---- GEMM1: h[256 x 64] = W1[256x384] @ msg[384 x 64] ----
  // acc[ct*2 + p] : row-tile (2*wv + p), column-tile ct
  v8f acc[8];
  #pragma unroll
  for (int i = 0; i < 8; ++i) acc[i] = zero8;

  const int m1a = (2 * wv)     * 16 + l16;   // A row, tile p=0
  const int m1b = (2 * wv + 1) * 16 + l16;   // A row, tile p=1

  for (int kc = 0; kc < 12; ++kc) {
    Frag af0, af1;
    #pragma unroll
    for (int v = 0; v < 8; ++v) {
      af0.u[v] = *(const unsigned int*)&w1h[m1a * K1 + 32 * kc + kA[v]];
      af1.u[v] = *(const unsigned int*)&w1h[m1b * K1 + 32 * kc + kA[v]];
    }
    #pragma unroll
    for (int ct = 0; ct < 4; ++ct) {
      Frag bf;
      #pragma unroll
      for (int v = 0; v < 8; ++v)
        bf.u[v] = *(const unsigned int*)&msg[(ct * 16 + l16) * MSG_LD + 32 * kc + kA[v]];
      acc[ct * 2 + 0] = __builtin_amdgcn_wmma_f32_16x16x32_f16(
          false, af0.v, false, bf.v, (short)0, acc[ct * 2 + 0], false, false);
      acc[ct * 2 + 1] = __builtin_amdgcn_wmma_f32_16x16x32_f16(
          false, af1.v, false, bf.v, (short)0, acc[ct * 2 + 1], false, false);
    }
  }

  // BN + ReLU, pack f16 pairs into hbuf. C/D layout: VGPR r -> M = r + 8*hh
  #pragma unroll
  for (int p = 0; p < 2; ++p) {
    int mb = (2 * wv + p) * 16 + 8 * hh;
    float sc[8], sh[8];
    #pragma unroll
    for (int r = 0; r < 8; ++r) { sc[r] = scale[mb + r]; sh[r] = shift[mb + r]; }
    #pragma unroll
    for (int ct = 0; ct < 4; ++ct) {
      int col = ct * 16 + l16;
      #pragma unroll
      for (int u = 0; u < 4; ++u) {
        float x0 = fmaxf(acc[ct * 2 + p][2 * u]     * sc[2 * u]     + sh[2 * u],     0.0f);
        float x1 = fmaxf(acc[ct * 2 + p][2 * u + 1] * sc[2 * u + 1] + sh[2 * u + 1], 0.0f);
        union { _Float16 h2[2]; unsigned int u32; } pk;
        pk.h2[0] = (_Float16)x0; pk.h2[1] = (_Float16)x1;
        *(unsigned int*)&hbuf[col * H_LD + mb + 2 * u] = pk.u32;
      }
    }
  }
  __syncthreads();

  // ---- GEMM2: up[128 x 64] = W2[128x256] @ h[256 x 64] ----
  v8f acc2[4];
  #pragma unroll
  for (int i = 0; i < 4; ++i) acc2[i] = zero8;

  const int m2 = wv * 16 + l16;  // A row for GEMM2 (wave owns one row-tile)

  for (int kc = 0; kc < 8; ++kc) {
    Frag af;
    #pragma unroll
    for (int v = 0; v < 8; ++v)
      af.u[v] = *(const unsigned int*)&w2h[m2 * M1 + 32 * kc + kA[v]];
    #pragma unroll
    for (int ct = 0; ct < 4; ++ct) {
      Frag bf;
      #pragma unroll
      for (int v = 0; v < 8; ++v)
        bf.u[v] = *(const unsigned int*)&hbuf[(ct * 16 + l16) * H_LD + 32 * kc + kA[v]];
      acc2[ct] = __builtin_amdgcn_wmma_f32_16x16x32_f16(
          false, af.v, false, bf.v, (short)0, acc2[ct], false, false);
    }
  }

  // Epilogue: + b2, scale by 1/cnt, atomic scatter-add into out (pre-init = ldesc)
  const int mb2 = wv * 16 + 8 * hh;
  float bias[8];
  #pragma unroll
  for (int r = 0; r < 8; ++r) bias[r] = b2[mb2 + r];
  #pragma unroll
  for (int ct = 0; ct < 4; ++ct) {
    int col = ct * 16 + l16;
    int   j  = jidx[col];
    float ci = cinv[col];
    #pragma unroll
    for (int r = 0; r < 8; ++r) {
      float val = (acc2[ct][r] + bias[r]) * ci;
      atomicAdd(&out[((size_t)b * DD + mb2 + r) * DN + j], val);
    }
  }
}

extern "C" void kernel_launch(void* const* d_in, const int* in_sizes, int n_in,
                              void* d_out, int out_size, void* d_ws, size_t ws_size,
                              hipStream_t stream) {
  const float* ldesc0 = (const float*)d_in[0];
  const float* ldesc1 = (const float*)d_in[1];
  const float* le0    = (const float*)d_in[2];
  const float* le1    = (const float*)d_in[3];
  const int*   idx0   = (const int*)d_in[4];
  const int*   idx1   = (const int*)d_in[5];
  const float* w1     = (const float*)d_in[6];
  const float* b1     = (const float*)d_in[7];
  const float* g      = (const float*)d_in[8];
  const float* beta   = (const float*)d_in[9];
  const float* mu     = (const float*)d_in[10];
  const float* var    = (const float*)d_in[11];
  const float* w2     = (const float*)d_in[12];
  const float* b2     = (const float*)d_in[13];

  char* ws = (char*)d_ws;
  unsigned int* counts = (unsigned int*)ws;                          // 2*B*N u32
  size_t off = (size_t)2 * DB * DN * 4;
  _Float16* w1h = (_Float16*)(ws + off);  off += (size_t)M1 * K1 * 2;
  _Float16* w2h = (_Float16*)(ws + off);  off += (size_t)M2 * M1 * 2;
  float* scl  = (float*)(ws + off);       off += M1 * 4;
  float* shf  = (float*)(ws + off);

  float* out0 = (float*)d_out;
  float* out1 = out0 + (size_t)DB * DD * DN;

  prep_kernel<<<96, 256, 0, stream>>>(w1, b1, g, beta, mu, var, w2, w1h, w2h, scl, shf);
  zero_counts<<<(2 * DB * DN + 255) / 256, 256, 0, stream>>>(counts);
  count_kernel<<<(DB * DE + 255) / 256, 256, 0, stream>>>(idx0, counts);
  count_kernel<<<(DB * DE + 255) / 256, 256, 0, stream>>>(idx1, counts + DB * DN);
  init_out<<<((DB * DD * DN) + 255) / 256, 256, 0, stream>>>(ldesc0, ldesc1, out0);

  dim3 grid(DE / COLS, DB);
  line_mp_kernel<<<grid, TPB, 0, stream>>>(ldesc0, le0, idx0, counts,
                                           w1h, w2h, scl, shf, b2, out0);
  line_mp_kernel<<<grid, TPB, 0, stream>>>(ldesc1, le1, idx1, counts + DB * DN,
                                           w1h, w2h, scl, shf, b2, out1);
}
